// Attention_85856396247881
// MI455X (gfx1250) — matrix operand order repaired and checked by reading, not compiled, not run on hardware
//
#include <hip/hip_runtime.h>
#include <hip/hip_bf16.h>

// ---------------------------------------------------------------------------
// Shapes (compile-time): B=1, S=2048, Q_DIM=KV_DIM=256, N_HEAD=8, C=32,
// OUT_DIM=256, N_AXIS=1.
// ---------------------------------------------------------------------------
#define S_LEN   2048
#define DIM     256
#define NHEAD   8
#define CDIM    32
#define OUTD    256

typedef __attribute__((ext_vector_type(16))) _Float16     v16h;
typedef __attribute__((ext_vector_type(8)))  float        v8f;
typedef __attribute__((ext_vector_type(4)))  unsigned int u32x4;
typedef __attribute__((ext_vector_type(8)))  int          i32x8;
typedef __attribute__((ext_vector_type(4)))  int          i32x4;

union Frag {                 // 32 bytes: one f16 WMMA A or B operand per lane
    v16h  v;
    u32x4 q[2];
};

// ---------------------------------------------------------------------------
// TDM: async-load one 16-row x 32-col f32 bias tile (2 KB) into LDS.
// D# group0: count=1 | lds_addr | global_addr[56:0] | type=2.
// D# group1: data_size=4B, tile 32x16, tensor_dim0=2048, dim1=16384,
//            dim0_stride=2048, no padding / multicast / iteration.
// Tracked on TENSORcnt; EXEC-independent (one issue per wave).
// ---------------------------------------------------------------------------
__device__ __forceinline__ void tdm_bias_tile(unsigned int lds_off,
                                              const float* gptr)
{
    unsigned long long ga = (unsigned long long)(uintptr_t)gptr;
    u32x4 g0 = { 1u,                                  // count=1, user mode
                 lds_off,                             // lds_addr (bytes)
                 (unsigned int)ga,                    // global_addr[31:0]
                 (unsigned int)((ga >> 32) & 0x01FFFFFFu) | (2u << 30) };
    i32x8 g1 = { 0x00020000,                          // data_size=2 (4 bytes)
                 0x08000000,                          // tensor_dim0[15:0]=2048
                 0x40000000,                          // tensor_dim1[15:0]=16384
                 0x00200000,                          // tile_dim0=32
                 16,                                  // tile_dim1=16
                 2048,                                // tensor_dim0_stride=2048
                 0, 0 };
    i32x4 gz = { 0, 0, 0, 0 };
#if defined(__clang_major__) && __clang_major__ >= 23
    i32x8 gz8 = { 0, 0, 0, 0, 0, 0, 0, 0 };
    __builtin_amdgcn_tensor_load_to_lds(g0, g1, gz, gz, gz8, 0);
#else
    __builtin_amdgcn_tensor_load_to_lds(g0, g1, gz, gz, 0);
#endif
}

// ---------------------------------------------------------------------------
// Kernel 1: fused QG / KV projections.
//   qg = X @ Wqg[h] + bqg   -> q (f16, pre-scaled by C^-1/2), g (f32)
//   kv = X @ Wkv[h] + bkv   -> k (f16 row-major), v (f16 transposed: [h][c][s])
// ---------------------------------------------------------------------------
__global__ __launch_bounds__(256) void proj_kernel(
    const float* __restrict__ qin, const float* __restrict__ kvin,
    const float* __restrict__ qgw, const float* __restrict__ kvw,
    const float* __restrict__ qgb, const float* __restrict__ kvb,
    _Float16* __restrict__ qf, _Float16* __restrict__ kf,
    _Float16* __restrict__ vT, float* __restrict__ gbuf)
{
    int idx   = blockIdx.x * 256 + threadIdx.x;      // 2^21 total
    int c     = idx & 63;                            // 0..63 (2C)
    int h     = (idx >> 6) & 7;
    int s     = (idx >> 9) & (S_LEN - 1);
    int which = idx >> 20;                           // 0 = qg, 1 = kv

    const float* x = (which ? kvin : qin) + (size_t)s * DIM;
    const float* w = (which ? kvw : qgw) + h * 64 + c;   // column, stride 512
    float acc = (which ? kvb : qgb)[h * 64 + c];
#pragma unroll 4
    for (int a = 0; a < DIM; ++a)
        acc = fmaf(x[a], w[(size_t)a * 512], acc);

    if (which == 0) {
        if (c < CDIM)
            qf[((size_t)(h * S_LEN + s)) * CDIM + c] =
                (_Float16)(acc * 0.17677669529663687f);   // * C^-1/2
        else
            gbuf[((size_t)(h * S_LEN + s)) * CDIM + (c - CDIM)] = acc;
    } else {
        if (c < CDIM)
            kf[((size_t)(h * S_LEN + s)) * CDIM + c] = (_Float16)acc;
        else
            vT[((size_t)(h * CDIM + (c - CDIM))) * S_LEN + s] = (_Float16)acc;
    }
}

// ---------------------------------------------------------------------------
// Kernel 2: o_weights (h,c,o) f32 -> transposed f16 (o, k=h*32+c) for WMMA B.
// ---------------------------------------------------------------------------
__global__ __launch_bounds__(256) void wprep_kernel(
    const float* __restrict__ ow, _Float16* __restrict__ owT)
{
    int o = blockIdx.x;      // 0..255
    int k = threadIdx.x;     // 0..255  (= h*32 + c)
    owT[(size_t)o * 256 + k] = (_Float16)ow[(size_t)k * OUTD + o];
}

// ---------------------------------------------------------------------------
// Kernel 3: flash attention per (head, 64-row query block).
// 4 waves / block; each wave owns a 16-row M tile and iterates 32-key tiles.
// Bias (the 128 MB roofline stream) arrives via double-buffered TDM tiles.
// A fragment (16x32 f16): lane = grp*16+m, bytes {grp*16, 32+grp*16} of row m.
// B fragment (32x16 f16): lane = grp*16+n, bytes {grp*16, 32+grp*16} (K-major).
// D/C tile: VGPR r, lanes 0-15 -> M=r, lanes 16-31 -> M=8+r, N = lane&15.
// ---------------------------------------------------------------------------
__global__ __launch_bounds__(128) void attn_kernel(
    const _Float16* __restrict__ qf, const _Float16* __restrict__ kf,
    const _Float16* __restrict__ vT, const float* __restrict__ gbuf,
    const float* __restrict__ bias, _Float16* __restrict__ aout)
{
    __shared__ __attribute__((aligned(16))) _Float16 Pbuf[4][16][32];
    __shared__ __attribute__((aligned(16))) float    Btile[4][2][16][32];

    const int lane = threadIdx.x & 31;
    const int wave = __builtin_amdgcn_readfirstlane((int)(threadIdx.x >> 5));
    const int l15  = lane & 15;
    const int grp  = lane >> 4;
    const int h    = blockIdx.y;
    const int qRow = blockIdx.x * 64 + wave * 16;

    const float* biasRow = bias + ((size_t)h * S_LEN + qRow) * S_LEN;
    const unsigned int ldsB0 = (unsigned int)(uintptr_t)&Btile[wave][0][0][0];
    const unsigned int ldsB1 = (unsigned int)(uintptr_t)&Btile[wave][1][0][0];

    // Kick off the first bias tile before anything else.
    tdm_bias_tile(ldsB0, biasRow);

    // Q fragment for this wave's 16 rows (loaded once, reused 128 times).
    Frag aq;
    {
        const char* base =
            (const char*)(qf + ((size_t)(h * S_LEN + qRow + l15)) * CDIM);
        aq.q[0] = *(const u32x4*)(base + grp * 16);
        aq.q[1] = *(const u32x4*)(base + 32 + grp * 16);
    }

    v8f O0 = {}, O1 = {};                 // output accumulators, c=[0,16),[16,32)
    float mrow[8], lrow[8];               // row max (uniform), per-LANE sum part
#pragma unroll
    for (int r = 0; r < 8; ++r) { mrow[r] = -1e30f; lrow[r] = 0.0f; }

    for (int k0 = 0; k0 < S_LEN; k0 += 32) {
        const int buf = (k0 >> 5) & 1;
        const bool more = (k0 + 32) < S_LEN;
        if (more)   // prefetch next bias tile into the other LDS buffer
            tdm_bias_tile(buf ? ldsB0 : ldsB1, biasRow + (k0 + 32));

        // ---- logits: two 16x16 WMMA tiles covering 32 keys ----
        v8f S0, S1;
        {
            Frag bk;
            v8f z = {};
            const char* kb0 =
                (const char*)(kf + ((size_t)(h * S_LEN + k0 + l15)) * CDIM);
            bk.q[0] = *(const u32x4*)(kb0 + grp * 16);
            bk.q[1] = *(const u32x4*)(kb0 + 32 + grp * 16);
            S0 = __builtin_amdgcn_wmma_f32_16x16x32_f16(
                     false, aq.v, false, bk.v, (short)0, z, false, false);
            const char* kb1 =
                (const char*)(kf + ((size_t)(h * S_LEN + k0 + 16 + l15)) * CDIM);
            bk.q[0] = *(const u32x4*)(kb1 + grp * 16);
            bk.q[1] = *(const u32x4*)(kb1 + 32 + grp * 16);
            S1 = __builtin_amdgcn_wmma_f32_16x16x32_f16(
                     false, aq.v, false, bk.v, (short)0, z, false, false);
        }

        // ---- consume current TDM bias tile (keep one in flight) ----
        if (more) __builtin_amdgcn_s_wait_tensorcnt(1);
        else      __builtin_amdgcn_s_wait_tensorcnt(0);
        asm volatile("" ::: "memory");

        const float* bt = &Btile[wave][buf][0][0];
#pragma unroll
        for (int r = 0; r < 8; ++r) {
            int row = r + 8 * grp;
            S0[r] += bt[row * 32 + l15];
            S1[r] += bt[row * 32 + 16 + l15];
        }

        // ---- online softmax (row max reduced; sum kept as per-lane partial) ----
#pragma unroll
        for (int r = 0; r < 8; ++r) {
            float t0 = S0[r], t1 = S1[r];
            float tm = fmaxf(t0, t1);
            tm = fmaxf(tm, __shfl_xor(tm, 8, 32));
            tm = fmaxf(tm, __shfl_xor(tm, 4, 32));
            tm = fmaxf(tm, __shfl_xor(tm, 2, 32));
            tm = fmaxf(tm, __shfl_xor(tm, 1, 32));
            float mNew = fmaxf(mrow[r], tm);
            float corr = __expf(mrow[r] - mNew);
            float p0 = __expf(t0 - mNew);
            float p1 = __expf(t1 - mNew);
            lrow[r] = lrow[r] * corr + (p0 + p1);   // deferred reduction
            mrow[r] = mNew;
            O0[r] *= corr;
            O1[r] *= corr;
            // stage P in f16, D-layout -> LDS (per-wave private tile)
            Pbuf[wave][r + 8 * grp][l15]      = (_Float16)p0;
            Pbuf[wave][r + 8 * grp][16 + l15] = (_Float16)p1;
        }
        asm volatile("s_wait_dscnt 0" ::: "memory");

        // ---- P fragment from LDS (A layout), V fragments from vT, PV WMMAs ----
        Frag ap;
        {
            const char* pb = (const char*)(&Pbuf[wave][l15][0]);
            ap.q[0] = *(const u32x4*)(pb + grp * 16);
            ap.q[1] = *(const u32x4*)(pb + 32 + grp * 16);
        }
        Frag bv;
        const char* vb0 =
            (const char*)(vT + ((size_t)(h * CDIM + l15)) * S_LEN + k0);
        bv.q[0] = *(const u32x4*)(vb0 + grp * 16);
        bv.q[1] = *(const u32x4*)(vb0 + 32 + grp * 16);
        O0 = __builtin_amdgcn_wmma_f32_16x16x32_f16(
                 false, ap.v, false, bv.v, (short)0, O0, false, false);
        const char* vb1 =
            (const char*)(vT + ((size_t)(h * CDIM + 16 + l15)) * S_LEN + k0);
        bv.q[0] = *(const u32x4*)(vb1 + grp * 16);
        bv.q[1] = *(const u32x4*)(vb1 + 32 + grp * 16);
        O1 = __builtin_amdgcn_wmma_f32_16x16x32_f16(
                 false, ap.v, false, bv.v, (short)0, O1, false, false);
    }

    // ---- finalize: reduce per-lane sums once, 1/l, sigmoid gate, store f16 ----
#pragma unroll
    for (int r = 0; r < 8; ++r) {
        float l = lrow[r];
        l += __shfl_xor(l, 8, 32);
        l += __shfl_xor(l, 4, 32);
        l += __shfl_xor(l, 2, 32);
        l += __shfl_xor(l, 1, 32);
        float inv = 1.0f / l;
        int row = qRow + r + 8 * grp;
        float g0 = gbuf[((size_t)(h * S_LEN) + row) * CDIM + l15];
        float g1 = gbuf[((size_t)(h * S_LEN) + row) * CDIM + 16 + l15];
        float s0 = 1.0f / (1.0f + __expf(-g0));
        float s1 = 1.0f / (1.0f + __expf(-g1));
        aout[(size_t)row * 256 + h * CDIM + l15]      = (_Float16)(O0[r] * inv * s0);
        aout[(size_t)row * 256 + h * CDIM + 16 + l15] = (_Float16)(O1[r] * inv * s1);
    }
}

// ---------------------------------------------------------------------------
// Kernel 4: output projection, K = heads*C = 256 fused into one WMMA GEMM.
// out[s, o] = sum_k aout[s, k] * owT[o, k] + o_bias[o]
// ---------------------------------------------------------------------------
__global__ __launch_bounds__(128) void outproj_kernel(
    const _Float16* __restrict__ aout, const _Float16* __restrict__ owT,
    const float* __restrict__ ob, float* __restrict__ out)
{
    const int lane = threadIdx.x & 31;
    const int wave = threadIdx.x >> 5;
    const int l15  = lane & 15;
    const int grp  = lane >> 4;
    const int m0   = blockIdx.x * 16;
    const int n0   = blockIdx.y * 64 + wave * 16;

    v8f acc = {};
    for (int kk = 0; kk < 256; kk += 32) {
        Frag a, b;
        const char* ab = (const char*)(aout + (size_t)(m0 + l15) * 256 + kk);
        a.q[0] = *(const u32x4*)(ab + grp * 16);
        a.q[1] = *(const u32x4*)(ab + 32 + grp * 16);
        const char* bb = (const char*)(owT + (size_t)(n0 + l15) * 256 + kk);
        b.q[0] = *(const u32x4*)(bb + grp * 16);
        b.q[1] = *(const u32x4*)(bb + 32 + grp * 16);
        acc = __builtin_amdgcn_wmma_f32_16x16x32_f16(
                  false, a.v, false, b.v, (short)0, acc, false, false);
    }
    float bo = ob[n0 + l15];         // o_bias shape (256, 1)
#pragma unroll
    for (int r = 0; r < 8; ++r)
        out[(size_t)(m0 + r + 8 * grp) * OUTD + n0 + l15] = acc[r] + bo;
}

// ---------------------------------------------------------------------------
// Host-side launcher (graph-capture safe: launches only, all on `stream`).
// Workspace layout (<= 6.2 MB):
//   [0,1MB)  q f16   [1,2MB) k f16   [2,3MB) vT f16
//   [3,5MB)  g f32   [5,6MB) aout f16   [6MB,+128KB) owT f16
// ---------------------------------------------------------------------------
extern "C" void kernel_launch(void* const* d_in, const int* in_sizes, int n_in,
                              void* d_out, int out_size, void* d_ws, size_t ws_size,
                              hipStream_t stream)
{
    (void)in_sizes; (void)n_in; (void)out_size; (void)ws_size;

    const float* qin  = (const float*)d_in[0];
    const float* kvin = (const float*)d_in[1];
    const float* bias = (const float*)d_in[2];
    const float* qgw  = (const float*)d_in[3];
    const float* kvw  = (const float*)d_in[4];
    const float* qgb  = (const float*)d_in[5];
    const float* kvb  = (const float*)d_in[6];
    const float* ow   = (const float*)d_in[7];
    const float* ob   = (const float*)d_in[8];
    float* out = (float*)d_out;

    char* ws = (char*)d_ws;
    _Float16* qf   = (_Float16*)(ws + (0u << 20));
    _Float16* kf   = (_Float16*)(ws + (1u << 20));
    _Float16* vT   = (_Float16*)(ws + (2u << 20));
    float*    gbuf = (float*)   (ws + (3u << 20));
    _Float16* aout = (_Float16*)(ws + (5u << 20));
    _Float16* owT  = (_Float16*)(ws + (6u << 20));

    proj_kernel<<<8192, 256, 0, stream>>>(qin, kvin, qgw, kvw, qgb, kvb,
                                          qf, kf, vT, gbuf);
    wprep_kernel<<<256, 256, 0, stream>>>(ow, owT);
    attn_kernel<<<dim3(32, 8), 128, 0, stream>>>(qf, kf, vT, gbuf, bias, aout);
    outproj_kernel<<<dim3(128, 4), 128, 0, stream>>>(aout, owT, ob, out);
}
